// DEGCN_51874615001205
// MI455X (gfx1250) — matrix operand order repaired and checked by reading, not compiled, not run on hardware
//
#include <hip/hip_runtime.h>
#include <hip/hip_bf16.h>
#include <stdint.h>

typedef __bf16 bf16_t;
typedef __attribute__((ext_vector_type(16))) __bf16 v16bf;
typedef __attribute__((ext_vector_type(8)))  __bf16 v8bf;
typedef __attribute__((ext_vector_type(8)))  float  v8f;

#define N_NODES 4096
#define BATCH   16
#define CDIM    16
#define ODIM    64
#define DDIM    10
#define CIN     80       // CDIM + ODIM
#define KCH     3
#define KTOT    2400     // DDIM*KCH*CIN
#define XC_LD   1280     // BATCH*CIN

static __device__ __forceinline__ bf16_t f2bf(float f) {
  unsigned u = __builtin_bit_cast(unsigned, f);
  unsigned r = (u + 0x7FFFu + ((u >> 16) & 1u)) >> 16;
  unsigned short h = (unsigned short)r;
  return __builtin_bit_cast(bf16_t, h);
}
static __device__ __forceinline__ float bf2f(bf16_t b) {
  unsigned short h = __builtin_bit_cast(unsigned short, b);
  unsigned u = ((unsigned)h) << 16;
  return __builtin_bit_cast(float, u);
}

// ---------------------------------------------------------------------------
// Prep: wpool [D,K,Cin,O] f32 -> BT [O, KTOT] bf16 (K-contiguous B-side)
// ---------------------------------------------------------------------------
__global__ __launch_bounds__(256) void prep_w_kernel(
    const float* __restrict__ gw, const float* __restrict__ uw,
    bf16_t* __restrict__ BTg, bf16_t* __restrict__ BTu)
{
  int idx = blockIdx.x * 256 + threadIdx.x;
  const int NG = KTOT * 128;
  if (idx < NG) {
    int o = idx / KTOT, t = idx - o * KTOT;
    BTg[idx] = f2bf(gw[t * 128 + o]);
  } else {
    int j = idx - NG;
    if (j < KTOT * 64) {
      int o = j / KTOT, t = j - o * KTOT;
      BTu[j] = f2bf(uw[t * 64 + o]);
    }
  }
}

// ---------------------------------------------------------------------------
// Prep: Xcat[n, b*80+i] = concat(X,H); also transposed copy, and X-part of C2
// ---------------------------------------------------------------------------
__global__ __launch_bounds__(256) void prep_xcat_kernel(
    const float* __restrict__ X, const float* __restrict__ H,
    bf16_t* __restrict__ Xcat, bf16_t* __restrict__ XcatT,
    bf16_t* __restrict__ C2,   bf16_t* __restrict__ C2T)
{
  int idx = blockIdx.x * 256 + threadIdx.x;   // over N_NODES * XC_LD
  int n = idx / XC_LD;
  int c = idx - n * XC_LD;
  int b = c / CIN;
  int i = c - b * CIN;
  float v = (i < CDIM) ? X[((size_t)b * N_NODES + n) * CDIM + i]
                       : H[((size_t)b * N_NODES + n) * ODIM + (i - CDIM)];
  bf16_t bv = f2bf(v);
  Xcat [(size_t)n * XC_LD + c] = bv;
  XcatT[(size_t)c * N_NODES + n] = bv;
  if (i < CDIM) {
    C2 [(size_t)n * XC_LD + c] = bv;
    C2T[(size_t)c * N_NODES + n] = bv;
  }
}

// ---------------------------------------------------------------------------
// S = softmax(relu(E E^T), axis=1); writes S (row-major) and S^T, bf16
// One block per row n; 256 threads x 16 columns each.
// ---------------------------------------------------------------------------
__global__ __launch_bounds__(256) void softmax_s_kernel(
    const float* __restrict__ E, bf16_t* __restrict__ S, bf16_t* __restrict__ ST)
{
  __shared__ float red[256];
  const int n = blockIdx.x;
  const int tid = threadIdx.x;
  float en[DDIM];
#pragma unroll
  for (int d = 0; d < DDIM; ++d) en[d] = E[n * DDIM + d];
  float lg[16];
  float lmax = 0.0f;                    // relu => logits >= 0
  const int m0 = tid * 16;
#pragma unroll
  for (int j = 0; j < 16; ++j) {
    int m = m0 + j;
    float dot = 0.0f;
#pragma unroll
    for (int d = 0; d < DDIM; ++d) dot += en[d] * E[m * DDIM + d];
    dot = dot > 0.0f ? dot : 0.0f;
    lg[j] = dot;
    lmax = fmaxf(lmax, dot);
  }
  red[tid] = lmax; __syncthreads();
  for (int s = 128; s > 0; s >>= 1) {
    if (tid < s) red[tid] = fmaxf(red[tid], red[tid + s]);
    __syncthreads();
  }
  float gmax = red[0];
  __syncthreads();
  float lsum = 0.0f;
#pragma unroll
  for (int j = 0; j < 16; ++j) { lg[j] = __expf(lg[j] - gmax); lsum += lg[j]; }
  red[tid] = lsum; __syncthreads();
  for (int s = 128; s > 0; s >>= 1) {
    if (tid < s) red[tid] += red[tid + s];
    __syncthreads();
  }
  float inv = 1.0f / red[0];
#pragma unroll
  for (int j = 0; j < 16; ++j) {
    int m = m0 + j;
    bf16_t v = f2bf(lg[j] * inv);
    S [(size_t)n * N_NODES + m] = v;
    ST[(size_t)m * N_NODES + n] = v;
  }
}

// ---------------------------------------------------------------------------
// GEMM: out[M x Ncols] = A[M x K] * B, BT given as [Ncols x K] (K-contiguous).
// bf16 in, f32 WMMA accumulate, bf16 out. t2mode: out = 2*C - I.
// Block = 8 waves; macro-tile 256x128; wave -> 32 rows x 128 cols
// (two A fragments per K-step; each B fragment feeds two WMMAs).
// ---------------------------------------------------------------------------
__global__ __launch_bounds__(256) void gemm_bf16_kernel(
    const bf16_t* __restrict__ A, const bf16_t* __restrict__ BT,
    bf16_t* __restrict__ out, int M, int Ncols, int K, int t2mode)
{
  const int lane = threadIdx.x & 31;
  const int wave = threadIdx.x >> 5;
  const int lrow = lane & 15;
  const int hi   = lane >> 4;
  const int row0 = blockIdx.y * 256 + wave * 32;
  const int col0 = blockIdx.x * 128;
  v8f acc0[8], acc1[8];
#pragma unroll
  for (int ct = 0; ct < 8; ++ct) {
    acc0[ct] = (v8f){0.f,0.f,0.f,0.f,0.f,0.f,0.f,0.f};
    acc1[ct] = (v8f){0.f,0.f,0.f,0.f,0.f,0.f,0.f,0.f};
  }
  const bf16_t* arow0 = A + (size_t)(row0 + lrow) * K;
  const bf16_t* arow1 = A + (size_t)(row0 + 16 + lrow) * K;
  for (int kb = 0; kb < K; kb += 32) {
    // 16-bit A fragment: lanes 0-15 carry K kb..+15; lanes 16-31
    // carry kb+16..+31 (contiguous, matching the B fragment).
    v8bf a00 = *(const v8bf*)(arow0 + kb + hi * 16);
    v8bf a01 = *(const v8bf*)(arow0 + kb + hi * 16 + 8);
    v8bf a10 = *(const v8bf*)(arow1 + kb + hi * 16);
    v8bf a11 = *(const v8bf*)(arow1 + kb + hi * 16 + 8);
    v16bf af0, af1;
#pragma unroll
    for (int j = 0; j < 8; ++j) {
      af0[j] = a00[j]; af0[8 + j] = a01[j];
      af1[j] = a10[j]; af1[8 + j] = a11[j];
    }
#pragma unroll
    for (int ct = 0; ct < 8; ++ct) {
      // B fragment: lane holds column (col0+ct*16+lrow), 16 contiguous K values.
      const bf16_t* bp = BT + (size_t)(col0 + ct * 16 + lrow) * K + kb + hi * 16;
      v16bf bfr = *(const v16bf*)bp;
      acc0[ct] = __builtin_amdgcn_wmma_f32_16x16x32_bf16(
          false, af0, false, bfr, (short)0, acc0[ct], false, false);
      acc1[ct] = __builtin_amdgcn_wmma_f32_16x16x32_bf16(
          false, af1, false, bfr, (short)0, acc1[ct], false, false);
    }
  }
#pragma unroll
  for (int ct = 0; ct < 8; ++ct) {
    int col = col0 + ct * 16 + lrow;
#pragma unroll
    for (int r = 0; r < 8; ++r) {
      int m0r = row0 + hi * 8 + r;        // C layout: VGPR r -> M = r + hi*8
      int m1r = m0r + 16;
      float v0 = acc0[ct][r];
      float v1 = acc1[ct][r];
      if (t2mode) {
        v0 = 2.0f * v0 - ((m0r == col) ? 1.0f : 0.0f);
        v1 = 2.0f * v1 - ((m1r == col) ? 1.0f : 0.0f);
      }
      out[(size_t)m0r * Ncols + col] = f2bf(v0);
      out[(size_t)m1r * Ncols + col] = f2bf(v1);
    }
  }
}

// ---------------------------------------------------------------------------
// Phase-C: per node n, [16(batch) x KTOT] * BT[O x KTOT] with A built on the
// fly as E[n,d] * XGk[n, b*80+i]. Fused GRU epilogues.
//   is_gate=1: sigmoid; cols 0..63 -> Z (write Z*H into C2/C2T), 64..127 -> R.
//   is_gate=0: tanh;  out = R*H + (1-R)*HC  (final output, f32 [B,N,O]).
// ---------------------------------------------------------------------------
__global__ __launch_bounds__(256) void phasec_kernel(
    const bf16_t* __restrict__ xg0, const bf16_t* __restrict__ xg1,
    const bf16_t* __restrict__ xg2, const bf16_t* __restrict__ BT,
    const float* __restrict__ E, const float* __restrict__ bpool,
    const float* __restrict__ H, float* __restrict__ Rbuf,
    bf16_t* __restrict__ C2, bf16_t* __restrict__ C2T,
    float* __restrict__ out, int is_gate)
{
  const int lane = threadIdx.x & 31;
  const int wave = threadIdx.x >> 5;
  const int lrow = lane & 15;             // batch row b (A side) / col (B side)
  const int hi   = lane >> 4;
  const int n = blockIdx.x * 8 + wave;
  const int O = is_gate ? 128 : 64;
  const int NCT = O >> 4;
  v8f acc[8];
#pragma unroll
  for (int ct = 0; ct < 8; ++ct) acc[ct] = (v8f){0.f,0.f,0.f,0.f,0.f,0.f,0.f,0.f};

  for (int kb = 0; kb < KTOT; kb += 32) {
    v16bf af;
#pragma unroll
    for (int half = 0; half < 2; ++half) {
      int t = kb + hi * 16 + half * 8;    // flat (d,k,i); 8-chunks never straddle
      int d = t / 240;
      int r = t - d * 240;
      int k = r / 80;
      int i = r - k * 80;
      const bf16_t* src = (k == 0) ? xg0 : ((k == 1) ? xg1 : xg2);
      v8bf c = *(const v8bf*)(src + (size_t)n * XC_LD + lrow * CIN + i);
      float s = E[n * DDIM + d];
#pragma unroll
      for (int j = 0; j < 8; ++j) af[half * 8 + j] = f2bf(s * bf2f(c[j]));
    }
#pragma unroll
    for (int ct = 0; ct < 8; ++ct) {
      if (ct < NCT) {
        const bf16_t* bp = BT + (size_t)(ct * 16 + lrow) * KTOT + kb + hi * 16;
        v16bf bfr = *(const v16bf*)bp;
        acc[ct] = __builtin_amdgcn_wmma_f32_16x16x32_bf16(
            false, af, false, bfr, (short)0, acc[ct], false, false);
      }
    }
  }

#pragma unroll
  for (int ct = 0; ct < 8; ++ct) {
    if (ct >= NCT) continue;
    int o = ct * 16 + lrow;
    float bias = 0.0f;
#pragma unroll
    for (int d = 0; d < DDIM; ++d) bias += E[n * DDIM + d] * bpool[d * O + o];
#pragma unroll
    for (int r = 0; r < 8; ++r) {
      int b = hi * 8 + r;
      float v = acc[ct][r] + bias;
      if (is_gate) {
        float sg = 1.0f / (1.0f + __expf(-v));
        if (o < ODIM) {                              // Z gate -> Z*H into C2
          float zh = sg * H[((size_t)b * N_NODES + n) * ODIM + o];
          bf16_t bv = f2bf(zh);
          int c = CDIM + o;
          C2 [(size_t)n * XC_LD + b * CIN + c] = bv;
          C2T[(size_t)(b * CIN + c) * N_NODES + n] = bv;
        } else {                                     // R gate
          Rbuf[((size_t)n * BATCH + b) * ODIM + (o - ODIM)] = sg;
        }
      } else {
        float hc = tanhf(v);
        float rg = Rbuf[((size_t)n * BATCH + b) * ODIM + o];
        float h  = H[((size_t)b * N_NODES + n) * ODIM + o];
        out[((size_t)b * N_NODES + n) * ODIM + o] = rg * h + (1.0f - rg) * hc;
      }
    }
  }
}

// ---------------------------------------------------------------------------
extern "C" void kernel_launch(void* const* d_in, const int* in_sizes, int n_in,
                              void* d_out, int out_size, void* d_ws, size_t ws_size,
                              hipStream_t stream)
{
  const float* X  = (const float*)d_in[0];
  const float* H  = (const float*)d_in[1];
  const float* E  = (const float*)d_in[2];
  const float* gw = (const float*)d_in[3];
  const float* gb = (const float*)d_in[4];
  const float* uw = (const float*)d_in[5];
  const float* ub = (const float*)d_in[6];
  float* out = (float*)d_out;

  char* base = (char*)d_ws;
  size_t off = 0;
  auto carve = [&](size_t bytes) -> void* {
    void* q = base + off;
    off += (bytes + 255) & ~(size_t)255;
    return q;
  };
  bf16_t* S     = (bf16_t*)carve((size_t)N_NODES * N_NODES * 2);
  bf16_t* ST    = (bf16_t*)carve((size_t)N_NODES * N_NODES * 2);
  bf16_t* T2    = (bf16_t*)carve((size_t)N_NODES * N_NODES * 2);
  bf16_t* Xcat  = (bf16_t*)carve((size_t)N_NODES * XC_LD * 2);
  bf16_t* XcatT = (bf16_t*)carve((size_t)N_NODES * XC_LD * 2);
  bf16_t* XG1   = (bf16_t*)carve((size_t)N_NODES * XC_LD * 2);
  bf16_t* XG2   = (bf16_t*)carve((size_t)N_NODES * XC_LD * 2);
  bf16_t* C2    = (bf16_t*)carve((size_t)N_NODES * XC_LD * 2);
  bf16_t* C2T   = (bf16_t*)carve((size_t)N_NODES * XC_LD * 2);
  bf16_t* BTg   = (bf16_t*)carve((size_t)128 * KTOT * 2);
  bf16_t* BTu   = (bf16_t*)carve((size_t)64 * KTOT * 2);
  float*  Rbuf  = (float*) carve((size_t)N_NODES * BATCH * ODIM * 4);
  (void)ws_size; (void)in_sizes; (void)n_in; (void)out_size;

  // 1) weight repack + input concat
  prep_w_kernel<<<(KTOT * 192) / 256, 256, 0, stream>>>(gw, uw, BTg, BTu);
  prep_xcat_kernel<<<(N_NODES * XC_LD) / 256, 256, 0, stream>>>(X, H, Xcat, XcatT, C2, C2T);

  // 2) supports: S (and S^T), then T2 = 2*S@S - I
  softmax_s_kernel<<<N_NODES, 256, 0, stream>>>(E, S, ST);
  gemm_bf16_kernel<<<dim3(32, 16), 256, 0, stream>>>(S, ST, T2, N_NODES, N_NODES, N_NODES, 1);

  // 3) gate GCN: graph conv then node-adaptive weight apply (fused sigmoid)
  gemm_bf16_kernel<<<dim3(XC_LD / 128, 16), 256, 0, stream>>>(S,  XcatT, XG1, N_NODES, XC_LD, N_NODES, 0);
  gemm_bf16_kernel<<<dim3(XC_LD / 128, 16), 256, 0, stream>>>(T2, XcatT, XG2, N_NODES, XC_LD, N_NODES, 0);
  phasec_kernel<<<N_NODES / 8, 256, 0, stream>>>(Xcat, XG1, XG2, BTg, E, gb, H, Rbuf, C2, C2T, nullptr, 1);

  // 4) update GCN on C = [X, Z*H], fused tanh + GRU combine into d_out
  gemm_bf16_kernel<<<dim3(XC_LD / 128, 16), 256, 0, stream>>>(S,  C2T, XG1, N_NODES, XC_LD, N_NODES, 0);
  gemm_bf16_kernel<<<dim3(XC_LD / 128, 16), 256, 0, stream>>>(T2, C2T, XG2, N_NODES, XC_LD, N_NODES, 0);
  phasec_kernel<<<N_NODES / 8, 256, 0, stream>>>(C2, XG1, XG2, BTu, E, ub, H, Rbuf, nullptr, nullptr, out, 0);
}
